// SwinTransformer_1228360647056
// MI455X (gfx1250) — matrix operand order repaired and checked by reading, not compiled, not run on hardware
//
#include <hip/hip_runtime.h>
#include <math.h>

// Problem constants (Swin window attention)
#define EMB   256
#define HEADS 8
#define HDIM  32
#define NP    192            // patches per window
#define WAVES 12             // one 16-row stripe per wave
#define TPB   (WAVES * 32)   // 384 threads, wave32

typedef __attribute__((ext_vector_type(16))) __bf16 v16bf;
typedef __attribute__((ext_vector_type(8)))  __bf16 v8bf;
typedef __attribute__((ext_vector_type(8)))  float  v8f;

// LDS layout (bf16 elements), rows padded to avoid bank conflicts
#define LDW 264   // W^T slice: [32][264]
#define LDQ 40    // Q, K: [192][40]
#define LDV 200   // V^T:  [32][200]
#define LDP 200   // P:    [192][200]

#define SW_OFF   0
#define SQ_OFF   (SW_OFF  + 32 * LDW)
#define SK_OFF   (SQ_OFF  + NP * LDQ)
#define SVT_OFF  (SK_OFF  + NP * LDQ)
#define SP_OFF   (SVT_OFF + 32 * LDV)
#define SMEM_ELEMS (SP_OFF + NP * LDP)   // 68608 elems -> 137216 bytes

__device__ __forceinline__ __bf16 f2bf(float x) {
  // round-to-nearest-even float -> bf16
  unsigned u = __builtin_bit_cast(unsigned, x);
  unsigned r = u + 0x7FFFu + ((u >> 16) & 1u);
  unsigned short hs = (unsigned short)(r >> 16);
  return __builtin_bit_cast(__bf16, hs);
}

// A-fragment (16x32, M x K) from row-major bf16 in LDS.
// ISA 16-bit A layout: lanes 0-15 -> rows, k in {0..7,16..23}; lanes 16-31 -> k in {8..15,24..31}.
__device__ __forceinline__ v16bf load_afrag_lds(const __bf16* p, int ld) {
  const int lane = threadIdx.x & 31;
  const __bf16* r = p + (lane & 15) * ld + ((lane >> 4) << 3);
  v8bf lo = *(const v8bf*)(r);
  v8bf hi = *(const v8bf*)(r + 16);
  return __builtin_shufflevector(lo, hi, 0,1,2,3,4,5,6,7,8,9,10,11,12,13,14,15);
}

// B-fragment (32x16, K x N) given row-major B^T ([n][k]) in LDS.
// ISA 16-bit B layout: lane -> column n; lanes 0-15 hold k in {0..7,16..23}, lanes 16-31 hold k in {8..15,24..31}.
__device__ __forceinline__ v16bf load_bfrag_lds(const __bf16* p, int ld) {
  const int lane = threadIdx.x & 31;
  const __bf16* r = p + (lane & 15) * ld + ((lane >> 4) << 3);
  v8bf lo = *(const v8bf*)(r);
  v8bf hi = *(const v8bf*)(r + 16);
  return __builtin_shufflevector(lo, hi, 0,1,2,3,4,5,6,7,8,9,10,11,12,13,14,15);
}

// A-fragment from row-major f32 in global memory, converting to bf16.
__device__ __forceinline__ v16bf load_afrag_g32(const float* p, int ld) {
  const int lane = threadIdx.x & 31;
  const float* r = p + (lane & 15) * ld + ((lane >> 4) << 3);
  v16bf f;
#pragma unroll
  for (int hh = 0; hh < 2; ++hh) {
    float4 a = *(const float4*)(r + hh * 16);
    float4 b = *(const float4*)(r + hh * 16 + 4);
    f[hh * 8 + 0] = f2bf(a.x); f[hh * 8 + 1] = f2bf(a.y);
    f[hh * 8 + 2] = f2bf(a.z); f[hh * 8 + 3] = f2bf(a.w);
    f[hh * 8 + 4] = f2bf(b.x); f[hh * 8 + 5] = f2bf(b.y);
    f[hh * 8 + 6] = f2bf(b.z); f[hh * 8 + 7] = f2bf(b.w);
  }
  return f;
}

__device__ __forceinline__ v8f wmma_bf16(v16bf a, v16bf b, v8f c) {
  return __builtin_amdgcn_wmma_f32_16x16x32_bf16(false, a, false, b, (short)0, c,
                                                 false, false);
}

// Stage one head's 256x32 weight slice, transposed+bf16, into LDS [32][LDW].
__device__ __forceinline__ void stageW(__bf16* sW, const float* W, int c0, int tid) {
  for (int idx = tid; idx < EMB * HDIM; idx += TPB) {
    int k = idx >> 5;   // 0..255 (input dim)
    int d = idx & 31;   // 0..31  (head output dim) -> contiguous global reads
    sW[d * LDW + k] = f2bf(W[k * EMB + c0 + d]);
  }
}

__global__ __launch_bounds__(TPB)
void swin_window_attn_fused(const float* __restrict__ patches,
                            const float* __restrict__ Wq, const float* __restrict__ bq,
                            const float* __restrict__ Wk, const float* __restrict__ bk,
                            const float* __restrict__ Wv, const float* __restrict__ bv,
                            const float* __restrict__ pos_bias,
                            float* __restrict__ out) {
  extern __shared__ __bf16 smem[];
  __bf16* sW  = smem + SW_OFF;
  __bf16* sQ  = smem + SQ_OFF;
  __bf16* sK  = smem + SK_OFF;
  __bf16* sVt = smem + SVT_OFF;
  __bf16* sP  = smem + SP_OFF;

  const int tid    = threadIdx.x;
  const int wave   = tid >> 5;
  const int lane   = tid & 31;
  const int col    = lane & 15;           // C/D-layout: column within tile
  const int rowoff = (lane >> 4) << 3;    // C/D-layout: +8 rows for hi half
  const int row0   = wave * 16;           // this wave's 16-row stripe

  const float* X = patches + (size_t)blockIdx.x * NP * EMB;
  float*       O = out     + (size_t)blockIdx.x * NP * EMB;

  // X stripe -> 8 A-fragments in registers; reused for all heads & projections.
  v16bf xf[8];
#pragma unroll
  for (int ks = 0; ks < 8; ++ks)
    xf[ks] = load_afrag_g32(X + row0 * EMB + ks * 32, EMB);

#pragma unroll 1
  for (int h = 0; h < HEADS; ++h) {
    const int c0 = h * HDIM;

    // ---------- Q = X @ Wq[:,c0:c0+32] + bq ----------
    stageW(sW, Wq, c0, tid);
    __syncthreads();
    {
      v8f a0 = {}, a1 = {};
#pragma unroll
      for (int ks = 0; ks < 8; ++ks) {
        a0 = wmma_bf16(xf[ks], load_bfrag_lds(sW + ks * 32, LDW), a0);
        a1 = wmma_bf16(xf[ks], load_bfrag_lds(sW + 16 * LDW + ks * 32, LDW), a1);
      }
      float b0 = bq[c0 + col], b1 = bq[c0 + 16 + col];
#pragma unroll
      for (int v = 0; v < 8; ++v) {
        int r = row0 + rowoff + v;
        sQ[r * LDQ + col]      = f2bf(a0[v] + b0);
        sQ[r * LDQ + 16 + col] = f2bf(a1[v] + b1);
      }
    }
    __syncthreads();

    // ---------- K ----------
    stageW(sW, Wk, c0, tid);
    __syncthreads();
    {
      v8f a0 = {}, a1 = {};
#pragma unroll
      for (int ks = 0; ks < 8; ++ks) {
        a0 = wmma_bf16(xf[ks], load_bfrag_lds(sW + ks * 32, LDW), a0);
        a1 = wmma_bf16(xf[ks], load_bfrag_lds(sW + 16 * LDW + ks * 32, LDW), a1);
      }
      float b0 = bk[c0 + col], b1 = bk[c0 + 16 + col];
#pragma unroll
      for (int v = 0; v < 8; ++v) {
        int r = row0 + rowoff + v;
        sK[r * LDQ + col]      = f2bf(a0[v] + b0);
        sK[r * LDQ + 16 + col] = f2bf(a1[v] + b1);
      }
    }
    __syncthreads();

    // ---------- V (stored transposed: sVt[d][p]) ----------
    stageW(sW, Wv, c0, tid);
    __syncthreads();
    {
      v8f a0 = {}, a1 = {};
#pragma unroll
      for (int ks = 0; ks < 8; ++ks) {
        a0 = wmma_bf16(xf[ks], load_bfrag_lds(sW + ks * 32, LDW), a0);
        a1 = wmma_bf16(xf[ks], load_bfrag_lds(sW + 16 * LDW + ks * 32, LDW), a1);
      }
      float b0 = bv[c0 + col], b1 = bv[c0 + 16 + col];
#pragma unroll
      for (int v = 0; v < 8; ++v) {
        int r = row0 + rowoff + v;
        sVt[col * LDV + r]        = f2bf(a0[v] + b0);
        sVt[(16 + col) * LDV + r] = f2bf(a1[v] + b1);
      }
    }
    __syncthreads();

    // ---------- S = Q K^T * (1/sqrt(EMB)) + pos_bias[h]; softmax ----------
    v16bf qf = load_afrag_lds(sQ + row0 * LDQ, LDQ);
    const float* pb = pos_bias + ((size_t)h * NP + row0 + rowoff) * NP + col;

    v8f s[12];
    float rmax[8];
#pragma unroll
    for (int v = 0; v < 8; ++v) rmax[v] = -3.0e38f;

#pragma unroll
    for (int j = 0; j < 12; ++j) {
      v8f c = {};
      c = wmma_bf16(qf, load_bfrag_lds(sK + j * 16 * LDQ, LDQ), c);
#pragma unroll
      for (int v = 0; v < 8; ++v) {
        float val = c[v] * 0.0625f + pb[v * NP + j * 16];
        s[j][v] = val;
        rmax[v] = fmaxf(rmax[v], val);
      }
    }
    // Row reduce over columns: row data for a VGPR lives in one 16-lane half.
#pragma unroll
    for (int v = 0; v < 8; ++v) {
      float m = rmax[v];
      m = fmaxf(m, __shfl_xor(m, 1, 32));
      m = fmaxf(m, __shfl_xor(m, 2, 32));
      m = fmaxf(m, __shfl_xor(m, 4, 32));
      m = fmaxf(m, __shfl_xor(m, 8, 32));
      rmax[v] = m;
    }
    float rsum[8];
#pragma unroll
    for (int v = 0; v < 8; ++v) rsum[v] = 0.0f;
#pragma unroll
    for (int j = 0; j < 12; ++j)
#pragma unroll
      for (int v = 0; v < 8; ++v) {
        float e = __expf(s[j][v] - rmax[v]);
        s[j][v] = e;
        rsum[v] += e;
      }
#pragma unroll
    for (int v = 0; v < 8; ++v) {
      float m = rsum[v];
      m += __shfl_xor(m, 1, 32);
      m += __shfl_xor(m, 2, 32);
      m += __shfl_xor(m, 4, 32);
      m += __shfl_xor(m, 8, 32);
      rsum[v] = m;
    }
    // Unnormalized P -> bf16 in LDS (own stripe), normalize after O GEMM.
#pragma unroll
    for (int j = 0; j < 12; ++j)
#pragma unroll
      for (int v = 0; v < 8; ++v)
        sP[(row0 + rowoff + v) * LDP + j * 16 + col] = f2bf(s[j][v]);
    __syncthreads();

    // ---------- O = P V ----------
    v8f o0 = {}, o1 = {};
#pragma unroll
    for (int ks = 0; ks < 6; ++ks) {
      v16bf pa = load_afrag_lds(sP + row0 * LDP + ks * 32, LDP);
      o0 = wmma_bf16(pa, load_bfrag_lds(sVt + ks * 32, LDV), o0);
      o1 = wmma_bf16(pa, load_bfrag_lds(sVt + 16 * LDV + ks * 32, LDV), o1);
    }
#pragma unroll
    for (int v = 0; v < 8; ++v) {
      float inv = 1.0f / rsum[v];
      size_t p = (size_t)(row0 + rowoff + v);
      O[p * EMB + c0 + col]      = o0[v] * inv;
      O[p * EMB + c0 + 16 + col] = o1[v] * inv;
    }
    __syncthreads();
  }
}

extern "C" void kernel_launch(void* const* d_in, const int* in_sizes, int n_in,
                              void* d_out, int out_size, void* d_ws, size_t ws_size,
                              hipStream_t stream) {
  const float* patches = (const float*)d_in[0];
  const float* Wq = (const float*)d_in[1];
  const float* bq = (const float*)d_in[2];
  const float* Wk = (const float*)d_in[3];
  const float* bk = (const float*)d_in[4];
  const float* Wv = (const float*)d_in[5];
  const float* bv = (const float*)d_in[6];
  const float* pb = (const float*)d_in[7];
  float* out = (float*)d_out;

  const int nwin = in_sizes[0] / (NP * EMB);   // B*NW = 512
  const size_t smem_bytes = (size_t)SMEM_ELEMS * sizeof(__bf16);  // ~134 KB

  hipLaunchKernelGGL(swin_window_attn_fused, dim3(nwin), dim3(TPB), smem_bytes,
                     stream, patches, Wq, bq, Wk, bk, Wv, bv, pb, out);
}
